// ViT4Block_53790170415805
// MI455X (gfx1250) — compile-verified, matrix operand order checked by reading, not run on hardware
//
#include <hip/hip_runtime.h>
#include <hip/hip_bf16.h>

typedef __attribute__((ext_vector_type(16))) _Float16 v16h;
typedef __attribute__((ext_vector_type(8)))  _Float16 v8h;
typedef __attribute__((ext_vector_type(8)))  float    v8f;

#define N_TOK   4096
#define EMBED   768
#define KPATCH  256
#define HEADS   12
#define HDIM    64

// ---------------------------------------------------------------------------
// WMMA helpers (CDNA5 f16 -> f32, 16x16x32)
// ---------------------------------------------------------------------------
__device__ __forceinline__ v8f wmma16(v16h a, v16h b, v8f c) {
    // (neg_a, A, neg_b, B, c_mod, C, reuse_a, reuse_b)
    return __builtin_amdgcn_wmma_f32_16x16x32_f16(false, a, false, b,
                                                  (short)0, c, false, false);
}

// Load a 16x32 fragment (A-layout; identical lane mapping works for B when
// the operand is stored [N][K] K-major).  ISA 7.12.2:
//   lanes 0-15 : row = lane,    K = [k0..k0+7,  k0+16..k0+23]
//   lanes 16-31: row = lane-16, K = [k0+8..+15, k0+24..k0+31]
// Works for both global and LDS operands (generic pointer; addrspace is
// inferred after inlining, giving global_load_b128 / ds_load_b128).
__device__ __forceinline__ v16h load_frag(const _Float16* base,
                                          int ld, int row0, int k0) {
    int lane = threadIdx.x & 31;
    int r    = lane & 15;
    int ks   = (lane & 16) ? 8 : 0;
    const _Float16* p = base + (size_t)(row0 + r) * ld + k0 + ks;
    v8h lo = *(const v8h*)(p);       // K: ks .. ks+7
    v8h hi = *(const v8h*)(p + 16);  // K: ks+16 .. ks+23
    v16h f;
#pragma unroll
    for (int i = 0; i < 8; ++i) { f[i] = lo[i]; f[8 + i] = hi[i]; }
    return f;
}

// ---------------------------------------------------------------------------
// Stage 0: patchify x -> f16 [4096][256], and f32 -> f16 weight conversions
// ---------------------------------------------------------------------------
__global__ void __launch_bounds__(256)
patchify_kernel(const float* __restrict__ x, _Float16* __restrict__ A) {
    int idx = blockIdx.x * 256 + threadIdx.x;       // 4096*256 total
    int t = idx >> 8, e = idx & 255;
    int py = t >> 6, px = t & 63;
    int i = e >> 4, j = e & 15;
    A[idx] = (_Float16)x[(size_t)(py * 16 + i) * 1024 + px * 16 + j];
}

__global__ void __launch_bounds__(256)
cvt_f32_f16_kernel(const float* __restrict__ src, _Float16* __restrict__ dst, int n) {
    int idx = blockIdx.x * 256 + threadIdx.x;
    if (idx < n) dst[idx] = (_Float16)src[idx];
}

// ---------------------------------------------------------------------------
// Stage 1: patch-embed GEMM (M=4096,N=768,K=256) + bias + LayerNorm -> f16 tok
// One block = 16 tokens; 8 waves x 6 N-tiles cover all 768 columns.
// ---------------------------------------------------------------------------
__global__ void __launch_bounds__(256)
patch_embed_ln_kernel(const _Float16* __restrict__ A,     // [4096][256]
                      const _Float16* __restrict__ W,     // [768][256]
                      const float* __restrict__ bias,     // [768]
                      const float* __restrict__ gamma,
                      const float* __restrict__ beta,
                      _Float16* __restrict__ tok)         // [4096][768]
{
    __shared__ __align__(16) float tokLds[16 * EMBED];    // 48 KB
    __shared__ float red[16 * 16 * 2];
    __shared__ float muS[16], rstdS[16];

    int tid  = threadIdx.x;
    int wave = tid >> 5;
    int lane = tid & 31;
    int m0   = blockIdx.x * 16;

    v8f zero = {};
    v8f acc[6];
#pragma unroll
    for (int t = 0; t < 6; ++t) acc[t] = zero;

    for (int k0 = 0; k0 < KPATCH; k0 += 32) {
        v16h af = load_frag(A, KPATCH, m0, k0);
#pragma unroll
        for (int t = 0; t < 6; ++t) {
            v16h bf = load_frag(W, KPATCH, wave * 96 + t * 16, k0);
            acc[t] = wmma16(af, bf, acc[t]);
        }
    }

    int half = lane >> 4, col = lane & 15;
#pragma unroll
    for (int t = 0; t < 6; ++t) {
        int n = wave * 96 + t * 16 + col;
        float b = bias[n];
#pragma unroll
        for (int r = 0; r < 8; ++r)
            tokLds[(r + half * 8) * EMBED + n] = acc[t][r] + b;
    }
    __syncthreads();

    // LayerNorm: 16 threads per row, 48 elements each
    int row = tid >> 4;
    int seg = tid & 15;
    float s = 0.f, ss = 0.f;
#pragma unroll 4
    for (int i = 0; i < 48; ++i) {
        float v = tokLds[row * EMBED + seg * 48 + i];
        s += v; ss += v * v;
    }
    red[(row * 16 + seg) * 2 + 0] = s;
    red[(row * 16 + seg) * 2 + 1] = ss;
    __syncthreads();
    if (seg == 0) {
        float S = 0.f, SS = 0.f;
        for (int i = 0; i < 16; ++i) {
            S  += red[(row * 16 + i) * 2 + 0];
            SS += red[(row * 16 + i) * 2 + 1];
        }
        float mu  = S * (1.f / EMBED);
        float var = SS * (1.f / EMBED) - mu * mu;
        muS[row]   = mu;
        rstdS[row] = rsqrtf(var + 1e-6f);
    }
    __syncthreads();
    float mu = muS[row], rs = rstdS[row];
#pragma unroll 4
    for (int i = 0; i < 48; ++i) {
        int n = seg * 48 + i;
        float v = (tokLds[row * EMBED + n] - mu) * rs * gamma[n] + beta[n];
        tok[(size_t)(m0 + row) * EMBED + n] = (_Float16)v;
    }
}

// ---------------------------------------------------------------------------
// Stage 2: QKV GEMM (M=4096,N=2304,K=768); scatter to per-head q,k,v layouts
// q: [12][4096][64] pre-scaled by d^-1/2;  k: [12][4096][64];  vT: [12][64][4096]
// ---------------------------------------------------------------------------
__global__ void __launch_bounds__(256)
qkv_gemm_kernel(const _Float16* __restrict__ X,   // [4096][768]
                const _Float16* __restrict__ W,   // [2304][768]
                _Float16* __restrict__ q,
                _Float16* __restrict__ k,
                _Float16* __restrict__ vT)
{
    int tid  = threadIdx.x;
    int wave = tid >> 5;
    int lane = tid & 31;
    int gt   = blockIdx.x * 8 + wave;             // 256*144 wave-tiles
    int m0   = (gt / 144) * 16;                   // consecutive waves share A
    int n0   = (gt % 144) * 16;

    v8f acc = {};
    for (int k0 = 0; k0 < EMBED; k0 += 32) {
        v16h af = load_frag(X, EMBED, m0, k0);
        v16h bf = load_frag(W, EMBED, n0, k0);
        acc = wmma16(af, bf, acc);
    }

    int half = lane >> 4, col = lane & 15;
    int n     = n0 + col;
    int which = n / EMBED;          // 0=q 1=k 2=v (tiles never straddle: 768%16==0)
    int e     = n % EMBED;
    int head  = e >> 6, dd = e & 63;
    const float qscale = 0.125f;    // 64^-0.5
#pragma unroll
    for (int r = 0; r < 8; ++r) {
        int m = m0 + r + half * 8;
        float v = acc[r];
        if (which == 0)
            q[((size_t)head * N_TOK + m) * HDIM + dd] = (_Float16)(v * qscale);
        else if (which == 1)
            k[((size_t)head * N_TOK + m) * HDIM + dd] = (_Float16)v;
        else
            vT[((size_t)head * HDIM + dd) * N_TOK + m] = (_Float16)v;
    }
}

// ---------------------------------------------------------------------------
// Stage 3: flash attention.  One block = 8 waves, ALL on the same head,
// covering 8 consecutive 16-row Q tiles (128 query rows).  Each 32-key
// K/V chunk is staged through LDS once per block (8x less L2 traffic),
// with global_prefetch for the next chunk.
// Per chunk per wave: S=QK^T (4 wmma) -> online softmax -> O+=PV (4 wmma)
// ---------------------------------------------------------------------------
__global__ void __launch_bounds__(256)
attn_kernel(const _Float16* __restrict__ q,   // [12][4096][64] (scaled)
            const _Float16* __restrict__ k,   // [12][4096][64]
            const _Float16* __restrict__ vT,  // [12][64][4096]
            float* __restrict__ out)          // [4096][768]
{
    __shared__ __align__(16) _Float16 kLds[32 * HDIM];    // 4 KB: 32 keys x 64
    __shared__ __align__(16) _Float16 vLds[HDIM * 32];    // 4 KB: 64 dims x 32 keys
    __shared__ __align__(16) _Float16 pbuf[8][16 * 32];   // 8 KB: per-wave P tiles

    int tid  = threadIdx.x;
    int wave = tid >> 5;
    int lane = tid & 31;

    int head = blockIdx.x >> 5;                // 12 heads x 32 blocks
    int m0   = ((blockIdx.x & 31) * 8 + wave) * 16;

    const _Float16* qh = q  + (size_t)head * N_TOK * HDIM;
    const _Float16* kh = k  + (size_t)head * N_TOK * HDIM;
    const _Float16* vh = vT + (size_t)head * HDIM * N_TOK;

    v16h qa0 = load_frag(qh, HDIM, m0, 0);
    v16h qa1 = load_frag(qh, HDIM, m0, 32);

    v8f zero = {};
    v8f o[4];
#pragma unroll
    for (int t = 0; t < 4; ++t) o[t] = zero;
    float mi[8], li[8];
#pragma unroll
    for (int r = 0; r < 8; ++r) { mi[r] = -1e30f; li[r] = 0.f; }

    int half = lane >> 4, col = lane & 15;
    _Float16* pb = pbuf[wave];

    // staging index maps: K chunk = 32 rows x 64 halfs, V chunk = 64 rows x 32
    int krow = tid >> 3, kseg = tid & 7;       // 256 thr x 8 halfs = 2048
    int vrow = tid >> 2, vseg = tid & 3;       // 256 thr x 8 halfs = 2048

    for (int j0 = 0; j0 < N_TOK; j0 += 32) {
        __syncthreads();                       // previous chunk fully consumed
        *(v8h*)(kLds + krow * HDIM + kseg * 8) =
            *(const v8h*)(kh + (size_t)(j0 + krow) * HDIM + kseg * 8);
        *(v8h*)(vLds + vrow * 32 + vseg * 8) =
            *(const v8h*)(vh + (size_t)vrow * N_TOK + j0 + vseg * 8);
        if (j0 + 32 < N_TOK) {                 // prefetch next chunk into WGP$
            __builtin_prefetch(kh + (size_t)(j0 + 32 + krow) * HDIM + kseg * 8, 0, 1);
            __builtin_prefetch(vh + (size_t)vrow * N_TOK + j0 + 32 + vseg * 8, 0, 1);
        }
        __syncthreads();                       // chunk visible to all waves

        // S tiles: keys [j0, j0+16) and [j0+16, j0+32)  (from LDS)
        v8f s0 = wmma16(qa0, load_frag(kLds, HDIM, 0,  0),  zero);
        s0     = wmma16(qa1, load_frag(kLds, HDIM, 0,  32), s0);
        v8f s1 = wmma16(qa0, load_frag(kLds, HDIM, 16, 0),  zero);
        s1     = wmma16(qa1, load_frag(kLds, HDIM, 16, 32), s1);

        // online softmax: each row lives in one 16-lane half (C-layout)
#pragma unroll
        for (int r = 0; r < 8; ++r) {
            float mx = fmaxf(s0[r], s1[r]);
            mx = fmaxf(mx, __shfl_xor(mx, 1, 32));
            mx = fmaxf(mx, __shfl_xor(mx, 2, 32));
            mx = fmaxf(mx, __shfl_xor(mx, 4, 32));
            mx = fmaxf(mx, __shfl_xor(mx, 8, 32));
            float mnew = fmaxf(mi[r], mx);
            float corr = __expf(mi[r] - mnew);
            float p0 = __expf(s0[r] - mnew);
            float p1 = __expf(s1[r] - mnew);
            float ps = p0 + p1;
            ps += __shfl_xor(ps, 1, 32);
            ps += __shfl_xor(ps, 2, 32);
            ps += __shfl_xor(ps, 4, 32);
            ps += __shfl_xor(ps, 8, 32);
            li[r] = li[r] * corr + ps;
            mi[r] = mnew;
            o[0][r] *= corr; o[1][r] *= corr; o[2][r] *= corr; o[3][r] *= corr;
            int m = r + half * 8;
            pb[m * 32 + col]      = (_Float16)p0;
            pb[m * 32 + 16 + col] = (_Float16)p1;
        }
        // re-fragment P (16x32) as A operand via LDS; same-wave DScnt ordering
        v16h pa = load_frag(pb, 32, 0, 0);
#pragma unroll
        for (int t = 0; t < 4; ++t) {
            v16h vb = load_frag(vLds, 32, t * 16, 0);    // N=dim, K=keys-in-chunk
            o[t] = wmma16(pa, vb, o[t]);
        }
    }

    // finalize: divide by row sums, write [4096][768] fp32
#pragma unroll
    for (int t = 0; t < 4; ++t) {
#pragma unroll
        for (int r = 0; r < 8; ++r) {
            int m = m0 + r + half * 8;
            int n = head * HDIM + t * 16 + col;
            out[(size_t)m * EMBED + n] = o[t][r] / li[r];
        }
    }
}

// ---------------------------------------------------------------------------
// Host orchestration
// ---------------------------------------------------------------------------
extern "C" void kernel_launch(void* const* d_in, const int* in_sizes, int n_in,
                              void* d_out, int out_size, void* d_ws, size_t ws_size,
                              hipStream_t stream) {
    const float* x      = (const float*)d_in[0];   // [1024*1024]
    const float* proj_w = (const float*)d_in[1];   // [768*256]
    const float* proj_b = (const float*)d_in[2];   // [768]
    const float* ln_g   = (const float*)d_in[3];   // [768]
    const float* ln_b   = (const float*)d_in[4];   // [768]
    const float* qkv_w  = (const float*)d_in[5];   // [2304*768]
    float* out = (float*)d_out;                    // [1*4096*768]

    char* ws = (char*)d_ws;
    size_t off = 0;
    auto carve = [&](size_t bytes) {
        char* p = ws + off;
        off = (off + bytes + 255) & ~(size_t)255;
        return p;
    };
    _Float16* Apat = (_Float16*)carve((size_t)N_TOK * KPATCH * 2);      // 2 MB
    _Float16* Wp   = (_Float16*)carve((size_t)EMBED * KPATCH * 2);      // 384 KB
    _Float16* Wq   = (_Float16*)carve((size_t)3 * EMBED * EMBED * 2);   // 3.5 MB
    _Float16* tok  = (_Float16*)carve((size_t)N_TOK * EMBED * 2);       // 6 MB
    _Float16* qb   = (_Float16*)carve((size_t)HEADS * N_TOK * HDIM * 2);
    _Float16* kb   = (_Float16*)carve((size_t)HEADS * N_TOK * HDIM * 2);
    _Float16* vTb  = (_Float16*)carve((size_t)HEADS * HDIM * N_TOK * 2);

    // Stage 0: patchify + weight conversion
    patchify_kernel<<<(N_TOK * KPATCH) / 256, 256, 0, stream>>>(x, Apat);
    cvt_f32_f16_kernel<<<(EMBED * KPATCH + 255) / 256, 256, 0, stream>>>(
        proj_w, Wp, EMBED * KPATCH);
    cvt_f32_f16_kernel<<<(3 * EMBED * EMBED + 255) / 256, 256, 0, stream>>>(
        qkv_w, Wq, 3 * EMBED * EMBED);

    // Stage 1: patch embed + LayerNorm
    patch_embed_ln_kernel<<<N_TOK / 16, 256, 0, stream>>>(
        Apat, Wp, proj_b, ln_g, ln_b, tok);

    // Stage 2: QKV GEMM
    qkv_gemm_kernel<<<(N_TOK / 16) * (3 * EMBED / 16) / 8, 256, 0, stream>>>(
        tok, Wq, qb, kb, vTb);

    // Stage 3: attention (8 waves per block share a head's K/V via LDS)
    attn_kernel<<<HEADS * (N_TOK / 16) / 8, 256, 0, stream>>>(qb, kb, vTb, out);
}
